// regressor_acf_46042049413650
// MI455X (gfx1250) — matrix operand order repaired and checked by reading, not compiled, not run on hardware
//
#include <hip/hip_runtime.h>
#include <hip/hip_bf16.h>
#include <stdint.h>

typedef __attribute__((ext_vector_type(16))) _Float16 v16h;
typedef __attribute__((ext_vector_type(8)))  _Float16 v8h;
typedef __attribute__((ext_vector_type(8)))  float    v8f;

// ---------------- problem constants ----------------
constexpr int SEQ_L   = 8192;   // sequence length
constexpr int KTOT    = 8224;   // contraction span (L + 32), multiple of 32
constexpr int FPAD    = 16;     // front zero pad of vp
constexpr int VP_LEN  = 16448;  // padded sequence in LDS (covers max B index 16415)
constexpr int POOLED  = 1024;
constexpr int L1N     = 512;
constexpr int L2N     = 256;
constexpr int OUTN    = 4;

// ---------------- fragment loaders ----------------
// A 16x32 f16 operand, this file's contiguous k order: lane<16 holds row M=lane, elems 0..15=K(kb..kb+15)
// contiguous; lanes>=16 are the +16 K-offset half (kb includes 16*hi).
static __device__ __forceinline__ v16h ldA_lds(const _Float16* p) {
  v16h r;
#pragma unroll
  for (int i = 0; i < 8; ++i) { r[i] = p[i]; r[i + 8] = p[8 + i]; }
  return r;
}
static __device__ __forceinline__ v16h ldA_aligned(const _Float16* p) {
  v8h lo = *(const v8h*)p;
  v8h hi = *(const v8h*)(p + 8);
  v16h r;
#pragma unroll
  for (int i = 0; i < 8; ++i) { r[i] = lo[i]; r[i + 8] = hi[i]; }
  return r;
}
// B 32x16 f16 operand, same contiguous k order as A: lane<16 = col N=lane holding K=0..15 contiguous,
// lanes>=16 hold K=16..31 (pointer already includes the 16*hi offset).
static __device__ __forceinline__ v16h ldB_lds16(const _Float16* p) {
  v8h lo = *(const v8h*)p;
  v8h hi = *(const v8h*)(p + 8);
  v16h r;
#pragma unroll
  for (int i = 0; i < 8; ++i) { r[i] = lo[i]; r[i + 8] = hi[i]; }
  return r;
}
static __device__ __forceinline__ v16h ldB_f32(const float* p) {
  v16h r;
#pragma unroll
  for (int i = 0; i < 16; ++i) r[i] = (_Float16)p[i];
  return r;
}
static __device__ __forceinline__ v16h ldB_f32_guard(const float* base, int off, bool valid) {
  v16h r;
#pragma unroll
  for (int i = 0; i < 16; ++i) r[i] = (_Float16)(valid ? base[off + i] : 0.0f);
  return r;
}

#define WMMA_F16(a, b, c) \
  __builtin_amdgcn_wmma_f32_16x16x32_f16(false, (a), false, (b), (short)0, (c), false, false)

// ============================================================================
// Kernel 1: per-row autocorrelation via WMMA tiles + pool8 + row-max normalize.
// grid = 256 rows, block = 256 threads (8 wave32). Wave w owns lag blocks 4w..4w+3.
// D[m,n] (block blk) = sum_t vp[t+16-m] * vp[t+16+16n+256*blk] = r[256*blk+16n+m]
// ============================================================================
__global__ __launch_bounds__(256) void acf_wmma_kernel(const float* __restrict__ x,
                                                       _Float16* __restrict__ normH) {
  // One buffer, two lives: first an f32 async-to-LDS staging copy of the row,
  // then (after register staging + barrier) the zero-padded f16 sequence.
  __shared__ __align__(16) unsigned char smem[VP_LEN * 2];
  __shared__ float pooledLds[POOLED];
  __shared__ float red[256];
  _Float16* vp   = (_Float16*)smem;
  float*    stage = (float*)smem;

  const int tid = threadIdx.x;
  const int row = blockIdx.x;
  const float* xr = x + (size_t)row * SEQ_L;

  // ---- Phase 0a: async-copy row (f32) straight into LDS, no VGPR round-trip.
  {
    const unsigned lds0 = (unsigned)(uintptr_t)smem;
#pragma unroll
    for (int k = 0; k < SEQ_L / (256 * 4); ++k) {       // 8 x b128 per thread
      const int idx = (k * 256 + tid) * 4;              // float index, 16B aligned
      const unsigned long long ga = (unsigned long long)(uintptr_t)(xr + idx);
      const unsigned la = lds0 + (unsigned)idx * 4u;    // LDS byte address
      asm volatile("global_load_async_to_lds_b128 %0, %1, off"
                   :: "v"(la), "v"(ga) : "memory");
    }
    asm volatile("s_wait_asynccnt 0x0" ::: "memory");
  }
  __syncthreads();

  // ---- Phase 0b: stage to registers, then rewrite as zero-padded f16 in place.
  float vals[SEQ_L / 256];
#pragma unroll
  for (int k = 0; k < SEQ_L / 256; ++k) vals[k] = stage[k * 256 + tid];
  __syncthreads();
#pragma unroll
  for (int k = 0; k < SEQ_L / 256; ++k) vp[FPAD + k * 256 + tid] = (_Float16)vals[k];
  for (int i = tid; i < VP_LEN; i += 256)
    if (i < FPAD || i >= FPAD + SEQ_L) vp[i] = (_Float16)0.0f;
  __syncthreads();

  const int wv   = tid >> 5;
  const int lane = tid & 31;
  const int lsub = lane & 15;
  const int hi   = lane >> 4;

  v8f acc[4] = {};
#pragma unroll 1
  for (int kc = 0; kc < KTOT / 32; ++kc) {
    const int t0 = kc * 32;
    const v16h a = ldA_lds(&vp[t0 + 16 * hi + FPAD - lsub]);
#pragma unroll
    for (int j = 0; j < 4; ++j) {
      const int blkg = wv * 4 + j;
      const v16h b = ldB_lds16(&vp[t0 + 16 * hi + FPAD + 16 * lsub + 256 * blkg]);
      acc[j] = WMMA_F16(a, b, acc[j]);
    }
  }

  // pool-by-8: D elem j in a lane covers lags 256*blk + 16*lsub + 8*hi + j
#pragma unroll
  for (int j = 0; j < 4; ++j) {
    const int blkg = wv * 4 + j;
    float pv = acc[j][0];
#pragma unroll
    for (int e = 1; e < 8; ++e) pv = fmaxf(pv, acc[j][e]);
    pooledLds[blkg * 32 + 2 * lsub + hi] = pv;
  }
  __syncthreads();

  // row max
  float m = -3.402823466e38f;
  for (int i = tid; i < POOLED; i += 256) m = fmaxf(m, pooledLds[i]);
  red[tid] = m;
  __syncthreads();
  for (int s = 128; s > 0; s >>= 1) {
    if (tid < s) red[tid] = fmaxf(red[tid], red[tid + s]);
    __syncthreads();
  }
  const float rmax = red[0];

  for (int i = tid; i < POOLED; i += 256)
    normH[(size_t)row * POOLED + i] = (_Float16)(pooledLds[i] / rmax);
}

// ============================================================================
// Kernel 2: MLP 1024->512->256->4 on 16 rows per block, all layers via WMMA.
// grid = 16 blocks (16 rows each), block = 256 threads (8 wave32).
// ============================================================================
__global__ __launch_bounds__(256) void mlp_wmma_kernel(
    const _Float16* __restrict__ normH,
    const float* __restrict__ W1, const float* __restrict__ b1,
    const float* __restrict__ W2, const float* __restrict__ b2,
    const float* __restrict__ W3, const float* __restrict__ b3,
    float* __restrict__ out) {
  __shared__ _Float16 h1[16 * L1N];
  __shared__ _Float16 h2[16 * L2N];

  const int tid  = threadIdx.x;
  const int wv   = tid >> 5;
  const int lane = tid & 31;
  const int lsub = lane & 15;
  const int hi   = lane >> 4;
  const int row0 = blockIdx.x * 16;

  // warm the L2-resident W1 stream for this wave's first B tile (global_prefetch_b8)
  __builtin_prefetch(W1 + (size_t)((wv * 4) * 16 + lsub) * POOLED + 16 * hi, 0, 0);

  // ---- layer 1: (16 x 1024) @ (1024 x 512), 32 n-tiles -> 4 per wave
  {
    v8f acc[4] = {};
    const _Float16* arow = normH + (size_t)(row0 + lsub) * POOLED;
#pragma unroll 1
    for (int kc = 0; kc < POOLED / 32; ++kc) {
      const int kb = kc * 32;
      const v16h a = ldA_aligned(arow + kb + 16 * hi);
#pragma unroll
      for (int j = 0; j < 4; ++j) {
        const int n = (wv * 4 + j) * 16 + lsub;
        const v16h b = ldB_f32(W1 + (size_t)n * POOLED + kb + 16 * hi);
        acc[j] = WMMA_F16(a, b, acc[j]);
      }
    }
#pragma unroll
    for (int j = 0; j < 4; ++j) {
      const int n = (wv * 4 + j) * 16 + lsub;
      const float bb = b1[n];
#pragma unroll
      for (int e = 0; e < 8; ++e) {
        const int mrow = e + 8 * hi;
        float v = acc[j][e] + bb;
        v = v > 0.0f ? v : 0.01f * v;
        h1[mrow * L1N + n] = (_Float16)v;
      }
    }
  }
  __syncthreads();

  // ---- layer 2: (16 x 512) @ (512 x 256), 16 n-tiles -> 2 per wave
  {
    v8f acc[2] = {};
#pragma unroll 1
    for (int kc = 0; kc < L1N / 32; ++kc) {
      const int kb = kc * 32;
      const v16h a = ldA_aligned(&h1[lsub * L1N + kb + 16 * hi]);
#pragma unroll
      for (int j = 0; j < 2; ++j) {
        const int n = (wv * 2 + j) * 16 + lsub;
        const v16h b = ldB_f32(W2 + (size_t)n * L1N + kb + 16 * hi);
        acc[j] = WMMA_F16(a, b, acc[j]);
      }
    }
#pragma unroll
    for (int j = 0; j < 2; ++j) {
      const int n = (wv * 2 + j) * 16 + lsub;
      const float bb = b2[n];
#pragma unroll
      for (int e = 0; e < 8; ++e) {
        const int mrow = e + 8 * hi;
        float v = acc[j][e] + bb;
        v = v > 0.0f ? v : 0.01f * v;
        h2[mrow * L2N + n] = (_Float16)v;
      }
    }
  }
  __syncthreads();

  // ---- layer 3: (16 x 256) @ (256 x 4), single tile (N padded to 16), wave 0
  if (wv == 0) {
    v8f acc = {};
#pragma unroll 1
    for (int kc = 0; kc < L2N / 32; ++kc) {
      const int kb = kc * 32;
      const v16h a = ldA_aligned(&h2[lsub * L2N + kb + 16 * hi]);
      const v16h b = ldB_f32_guard(W3, lsub * L2N + kb + 16 * hi, lsub < OUTN);
      acc = WMMA_F16(a, b, acc);
    }
    if (lsub < OUTN) {
      const float bb = b3[lsub];
#pragma unroll
      for (int e = 0; e < 8; ++e) {
        const int mrow = e + 8 * hi;
        out[(size_t)(row0 + mrow) * OUTN + lsub] = acc[e] + bb;
      }
    }
  }
}

// ============================================================================
extern "C" void kernel_launch(void* const* d_in, const int* in_sizes, int n_in,
                              void* d_out, int out_size, void* d_ws, size_t ws_size,
                              hipStream_t stream) {
  (void)in_sizes; (void)n_in; (void)out_size; (void)ws_size;
  const float* x  = (const float*)d_in[0];
  const float* W1 = (const float*)d_in[1];
  const float* b1 = (const float*)d_in[2];
  const float* W2 = (const float*)d_in[3];
  const float* b2 = (const float*)d_in[4];
  const float* W3 = (const float*)d_in[5];
  const float* b3 = (const float*)d_in[6];
  float* out = (float*)d_out;

  _Float16* normH = (_Float16*)d_ws;  // 256 x 1024 f16 = 512 KB scratch

  acf_wmma_kernel<<<256, 256, 0, stream>>>(x, normH);
  mlp_wmma_kernel<<<16, 256, 0, stream>>>(normH, W1, b1, W2, b2, W3, b3, out);
}
